// EUNN_79783312491079
// MI455X (gfx1250) — matrix operand order repaired and checked by reading, not compiled, not run on hardware
//
#include <hip/hip_runtime.h>
#include <hip/hip_bf16.h>

// ---------------------------------------------------------------------------
// EUNN: 256 layers x (2 Givens stages + rolls) over (4096, 1024) complex.
// The scan is complex-linear => out = x * M. With interleaved (re,im) storage
// this is one real SGEMM:  O(4096x2048) = X(4096x2048) * W(2048x2048).
// W is built once (25% of the GEMM FLOPs), then the GEMM runs as an
// f16 split-2 (3-product) GEMM on V_WMMA_F32_16X16X32_F16:
//     X = Xh + Xl,  W = Wh + Wl   (f16 hi + f16 residual)
//     O ~= Xh*Wh + Xh*Wl + Xl*Wh      (f32 accumulate, err ~1e-5 rel)
// 8x FLOP/instruction vs the f32 16x16x4 WMMA path.
// ---------------------------------------------------------------------------

typedef __attribute__((ext_vector_type(2)))  float    v2f;
typedef __attribute__((ext_vector_type(8)))  float    v8f;
typedef __attribute__((ext_vector_type(4)))  _Float16 v4h;
typedef __attribute__((ext_vector_type(8)))  _Float16 v8h;
typedef __attribute__((ext_vector_type(16))) _Float16 v16h;

#define H   1024
#define H2  512
#define C2  256
#define KGEMM 2048

__device__ __forceinline__ void split2(float v, _Float16& hi, _Float16& lo) {
  hi = (_Float16)v;
  lo = (_Float16)(v - (float)hi);
}

// ---------------------------------------------------------------------------
// Kernel 0: pack rotation coefficients.
// For pair i, step c, stage s: a=cp*ct, b=sp*ct, c=ct, s=st, d=cp*st, e=sp*st.
// Even-elem update: x' = (a,b)*xe + (-d,-e)*xo ; odd: x' = c*xo + s*xe (real).
// Layout: coeff[((c*2+s)*512 + i)*6 + k]
// ---------------------------------------------------------------------------
__global__ __launch_bounds__(256) void eunn_coeff(
    const float* __restrict__ phi0, const float* __restrict__ theta0,
    const float* __restrict__ phi1, const float* __restrict__ theta1,
    float* __restrict__ coeff) {
  int idx = blockIdx.x * blockDim.x + threadIdx.x;   // over C2*H2
  if (idx >= C2 * H2) return;
  int c = idx >> 9;          // 0..255
  int i = idx & 511;         // 0..511
  const float* ph[2] = {phi0, phi1};
  const float* th[2] = {theta0, theta1};
#pragma unroll
  for (int s = 0; s < 2; ++s) {
    float p = ph[s][i * C2 + c];
    float t = th[s][i * C2 + c];
    float sp, cp, st, ct;
    sincosf(p, &sp, &cp);
    sincosf(t, &st, &ct);
    float* o = coeff + (size_t)((c * 2 + s) * 512 + i) * 6;
    o[0] = cp * ct;  o[1] = sp * ct;  o[2] = ct;
    o[3] = st;       o[4] = cp * st;  o[5] = sp * st;
  }
}

// ---------------------------------------------------------------------------
// Kernel 1: build W by running the scan on basis vectors e_j; emit W
// TRANSPOSED and split to f16 hi/lo:  WhT/WlT[n][k], n,k in [0,2048).
//   W[2j][2k]=Re M, W[2j][2k+1]=Im M, W[2j+1][2k]=-Im M, W[2j+1][2k+1]=Re M
// Block g owns basis rows j = 4g..4g+3; state in LDS (32 KB).
// Each thread owns pair indices {tid, tid+256}; coefficients are read from
// L2 into registers once per stage and reused across the 4 rows.
// Stage 0 couples (2i,2i+1); roll(-1)+stage+roll(+1) == couple
// (2i+1, (2i+2) mod 1024) in place.
// ---------------------------------------------------------------------------
__device__ __forceinline__ void givens_update(
    float (*st)[H][2], int r, int e0, int e1, const float* cf) {
  float a = cf[0], b = cf[1], cc = cf[2], ss = cf[3], d = cf[4], e = cf[5];
  float er = st[r][e0][0], ei = st[r][e0][1];
  float pr = st[r][e1][0], pi = st[r][e1][1];
  float ner = fmaf(a, er, fmaf(-b, ei, fmaf(-d, pr, e * pi)));
  float nei = fmaf(a, ei, fmaf( b, er, fmaf(-d, pi, -e * pr)));
  float npr = fmaf(cc, pr, ss * er);
  float npi = fmaf(cc, pi, ss * ei);
  st[r][e0][0] = ner; st[r][e0][1] = nei;
  st[r][e1][0] = npr; st[r][e1][1] = npi;
}

__global__ __launch_bounds__(256) void eunn_build_w(
    const float* __restrict__ coeff,
    _Float16* __restrict__ WhT, _Float16* __restrict__ WlT) {
  __shared__ float st[4][H][2];      // 32 KB
  const int tid = threadIdx.x;
  const int g = blockIdx.x;          // 0..255
  const int i0 = tid, i1 = tid + 256;

  for (int k = tid; k < 4 * H * 2; k += 256) ((float*)st)[k] = 0.0f;
  __syncthreads();
  if (tid < 4) st[tid][4 * g + tid][0] = 1.0f;

  for (int c = 0; c < C2; ++c) {
    // ---- stage 0: pairs (2i, 2i+1)
    const float* c00 = coeff + ((size_t)(c * 2) * 512 + i0) * 6;
    const float* c01 = coeff + ((size_t)(c * 2) * 512 + i1) * 6;
    float A0[6], A1[6];
#pragma unroll
    for (int k = 0; k < 6; ++k) { A0[k] = c00[k]; A1[k] = c01[k]; }
    __syncthreads();
#pragma unroll
    for (int r = 0; r < 4; ++r) {
      givens_update(st, r, 2 * i0, 2 * i0 + 1, A0);
      givens_update(st, r, 2 * i1, 2 * i1 + 1, A1);
    }
    // ---- stage 1: pairs (2i+1, (2i+2) mod 1024)
    const float* c10 = coeff + ((size_t)(c * 2 + 1) * 512 + i0) * 6;
    const float* c11 = coeff + ((size_t)(c * 2 + 1) * 512 + i1) * 6;
#pragma unroll
    for (int k = 0; k < 6; ++k) { A0[k] = c10[k]; A1[k] = c11[k]; }
    __syncthreads();
#pragma unroll
    for (int r = 0; r < 4; ++r) {
      givens_update(st, r, 2 * i0 + 1, (2 * i0 + 2) & (H - 1), A0);
      givens_update(st, r, 2 * i1 + 1, (2 * i1 + 2) & (H - 1), A1);
    }
  }
  __syncthreads();

  // Emit transposed, f16-split W. Block g owns WT columns 8g..8g+7; each
  // thread packs one full WT row chunk (8 f16 = 16B) per iteration.
  for (int q = tid; q < 2048; q += 256) {
    int k_ = q >> 1, b = q & 1;
    __attribute__((aligned(16))) _Float16 vh[8];
    __attribute__((aligned(16))) _Float16 vl[8];
#pragma unroll
    for (int r = 0; r < 4; ++r) {
      float re = st[r][k_][0], im = st[r][k_][1];
      float v0 = b ? im : re;          // WT[q][2j]
      float v1 = b ? re : -im;         // WT[q][2j+1]
      split2(v0, vh[2 * r],     vl[2 * r]);
      split2(v1, vh[2 * r + 1], vl[2 * r + 1]);
    }
    *(float4*)&WhT[(size_t)q * KGEMM + 8 * g] = *(const float4*)vh;
    *(float4*)&WlT[(size_t)q * KGEMM + 8 * g] = *(const float4*)vl;
  }
}

// ---------------------------------------------------------------------------
// Kernel 2: O = X*W as f16 split-2 GEMM on V_WMMA_F32_16X16X32_F16.
// Block tile 128(M) x 128(N), 8 waves (2x4), wave tile 64x32 = 4x2 tiles,
// K-chunk 32 (one WMMA K-depth): 24 WMMAs/chunk/wave.
// A frag (16x32 f16): lane=M; K = {8h..8h+7} U {16+8h..16+8h+7}  -> 2x v8h
// B frag (32x16 f16): lane=N; K = {8h..8h+7} U {16+8h..16+8h+7}                   -> 2x v8h
// C/D (16x16 f32):    lane=N; M = vgpr + 8h
// LDS rows padded to 40 f16 (80 B): 16B-aligned rows, optimal bank spread.
// ---------------------------------------------------------------------------
__device__ __forceinline__ v16h cat8(v8h a, v8h b) {
  return __builtin_shufflevector(a, b, 0, 1, 2, 3, 4, 5, 6, 7,
                                        8, 9, 10, 11, 12, 13, 14, 15);
}

__global__ __launch_bounds__(256) void eunn_gemm(
    const float* __restrict__ X,
    const _Float16* __restrict__ WhT, const _Float16* __restrict__ WlT,
    float* __restrict__ O) {
  __shared__ _Float16 Xsh[128][40];   // 10 KB each
  __shared__ _Float16 Xsl[128][40];
  __shared__ _Float16 Wsh[128][40];   // [N][K] (transposed in global already)
  __shared__ _Float16 Wsl[128][40];

  const int tid  = threadIdx.x;
  const int lane = tid & 31;
  const int wave = tid >> 5;
  const int lm = lane & 15;        // M for A, N for B/C
  const int h  = lane >> 4;        // half select
  const int M0 = blockIdx.x * 128;
  const int N0 = blockIdx.y * 128;
  const int wm = (wave & 1) * 64;
  const int wn = (wave >> 1) * 32;

  v8f acc[4][2];
#pragma unroll
  for (int mi = 0; mi < 4; ++mi)
#pragma unroll
    for (int ni = 0; ni < 2; ++ni) acc[mi][ni] = (v8f){0,0,0,0,0,0,0,0};

  for (int kc = 0; kc < KGEMM; kc += 32) {
    // ---- stage X tile (128x32 f32 -> f16 hi/lo): 4 float4 per thread
#pragma unroll
    for (int it = 0; it < 4; ++it) {
      int idx = tid + it * 256;              // 0..1023
      int r = idx >> 3, c4 = (idx & 7) * 4;
      float4 v = *(const float4*)&X[(size_t)(M0 + r) * KGEMM + kc + c4];
      _Float16 h0, l0, h1, l1, h2, l2, h3, l3;
      split2(v.x, h0, l0); split2(v.y, h1, l1);
      split2(v.z, h2, l2); split2(v.w, h3, l3);
      v4h th = {h0, h1, h2, h3};
      v4h tl = {l0, l1, l2, l3};
      *(v4h*)&Xsh[r][c4] = th;
      *(v4h*)&Xsl[r][c4] = tl;
    }
    // ---- stage W tile (128 cols x 32 K, f16, already transposed+split)
#pragma unroll
    for (int it = 0; it < 2; ++it) {
      int idx = tid + it * 256;              // 0..511
      int col = idx >> 2, k8 = (idx & 3) * 8;
      *(v8h*)&Wsh[col][k8] =
          *(const v8h*)&WhT[(size_t)(N0 + col) * KGEMM + kc + k8];
      *(v8h*)&Wsl[col][k8] =
          *(const v8h*)&WlT[(size_t)(N0 + col) * KGEMM + kc + k8];
    }
    __syncthreads();

    v16h Bh[2], Bl[2];
#pragma unroll
    for (int ni = 0; ni < 2; ++ni) {
      int col = wn + ni * 16 + lm;
      Bh[ni] = cat8(*(const v8h*)&Wsh[col][8 * h],
                    *(const v8h*)&Wsh[col][16 + 8 * h]);
      Bl[ni] = cat8(*(const v8h*)&Wsl[col][8 * h],
                    *(const v8h*)&Wsl[col][16 + 8 * h]);
    }
#pragma unroll
    for (int mi = 0; mi < 4; ++mi) {
      int row = wm + mi * 16 + lm;
      v16h Ah = cat8(*(const v8h*)&Xsh[row][8 * h],
                     *(const v8h*)&Xsh[row][16 + 8 * h]);
      v16h Al = cat8(*(const v8h*)&Xsl[row][8 * h],
                     *(const v8h*)&Xsl[row][16 + 8 * h]);
#pragma unroll
      for (int ni = 0; ni < 2; ++ni) {
        // low-order products first, then the dominant one
        acc[mi][ni] = __builtin_amdgcn_wmma_f32_16x16x32_f16(
            false, Al, false, Bh[ni], (short)0, acc[mi][ni], false, false);
        acc[mi][ni] = __builtin_amdgcn_wmma_f32_16x16x32_f16(
            false, Ah, false, Bl[ni], (short)0, acc[mi][ni], false, false);
        acc[mi][ni] = __builtin_amdgcn_wmma_f32_16x16x32_f16(
            false, Ah, false, Bh[ni], (short)0, acc[mi][ni], false, false);
      }
    }
    __syncthreads();
  }

  // writeback: within each 16x16 tile, M = r + 8h, N = lm
#pragma unroll
  for (int mi = 0; mi < 4; ++mi) {
#pragma unroll
    for (int ni = 0; ni < 2; ++ni) {
      int col  = N0 + wn + ni * 16 + lm;
      int rowb = M0 + wm + mi * 16 + 8 * h;
#pragma unroll
      for (int r = 0; r < 8; ++r)
        O[(size_t)(rowb + r) * KGEMM + col] = acc[mi][ni][r];
    }
  }
}

// ---------------------------------------------------------------------------
extern "C" void kernel_launch(void* const* d_in, const int* in_sizes, int n_in,
                              void* d_out, int out_size, void* d_ws, size_t ws_size,
                              hipStream_t stream) {
  const float* x      = (const float*)d_in[0];   // (4096, 1024, 2)
  const float* phi0   = (const float*)d_in[1];   // (512, 256)
  const float* theta0 = (const float*)d_in[2];
  const float* phi1   = (const float*)d_in[3];
  const float* theta1 = (const float*)d_in[4];
  float* out = (float*)d_out;                    // (4096, 1024, 2)

  _Float16* WhT  = (_Float16*)d_ws;                        // 8 MB
  _Float16* WlT  = WhT + (size_t)KGEMM * KGEMM;            // 8 MB
  float*    coeff = (float*)(WlT + (size_t)KGEMM * KGEMM); // 6 MB

  eunn_coeff<<<dim3((C2 * H2 + 255) / 256), dim3(256), 0, stream>>>(
      phi0, theta0, phi1, theta1, coeff);
  eunn_build_w<<<dim3(256), dim3(256), 0, stream>>>(coeff, WhT, WlT);
  eunn_gemm<<<dim3(4096 / 128, 2048 / 128), dim3(256), 0, stream>>>(
      x, WhT, WlT, out);

  (void)in_sizes; (void)n_in; (void)out_size; (void)ws_size;
}
